// MuleDetectorGNN_27642409517219
// MI455X (gfx1250) — matrix operand order repaired and checked by reading, not compiled, not run on hardware
//
#include <hip/hip_runtime.h>

// ---------------------------------------------------------------------------
// GraphSAGE 3-layer GNN for MI455X (gfx1250, wave32, WMMA).
// Strategy: edge aggregation (L2-resident float atomics) -> fused
// [mean|x] @ [Wl|Wr]^T GEMM via v_wmma_f32_16x16x32_bf16 -> BN+ReLU -> repeat.
// ---------------------------------------------------------------------------

typedef __bf16 bf16_t;
typedef bf16_t v16bf __attribute__((ext_vector_type(16)));
typedef float  v8f   __attribute__((ext_vector_type(8)));

__device__ __forceinline__ unsigned short f2bf(float f) {
    // round-to-nearest-even f32 -> bf16
    unsigned u = __float_as_uint(f);
    unsigned r = u + 0x7FFFu + ((u >> 16) & 1u);
    return (unsigned short)(r >> 16);
}

// ----------------------------- degree -------------------------------------
__global__ __launch_bounds__(256) void degree_kernel(
    const int* __restrict__ dst, float* __restrict__ deg, int E) {
    int e = blockIdx.x * blockDim.x + threadIdx.x;
    if (e < E) atomicAdd(&deg[dst[e]], 1.0f);
}

// --------------------------- edge aggregation ------------------------------
// One thread per (edge, 8-feature chunk): gather x[src], atomic add into agg[dst].
__global__ __launch_bounds__(256) void aggregate_kernel(
    const float* __restrict__ x, const int* __restrict__ src,
    const int* __restrict__ dst, float* __restrict__ agg,
    int E, int F, int nchunk) {
    int idx = blockIdx.x * blockDim.x + threadIdx.x;
    int e = idx / nchunk;
    if (e >= E) return;
    int c = idx - e * nchunk;
    int s = src[e], d = dst[e];
    const float* xs = x + (size_t)s * F;
    float* ad = agg + (size_t)d * F;
    int f0 = c * 8;
    int f1 = (f0 + 8 < F) ? f0 + 8 : F;
    for (int f = f0; f < f1; ++f) atomicAdd(&ad[f], xs[f]);
}

// ------------------- build bf16 concatenated features ----------------------
// Xcat[n][k] = (k<F) ? agg[n][k]/max(deg,1) : (k<2F) ? x[n][k-F] : 0
__global__ __launch_bounds__(256) void build_xcat_kernel(
    const float* __restrict__ agg, const float* __restrict__ deg,
    const float* __restrict__ x, unsigned short* __restrict__ Xcat,
    int N, int F, int Kpad) {
    int idx = blockIdx.x * blockDim.x + threadIdx.x;
    if (idx >= N * Kpad) return;
    int n = idx / Kpad;
    int k = idx - n * Kpad;
    float v = 0.0f;
    if (k < F)          v = agg[(size_t)n * F + k] / fmaxf(deg[n], 1.0f);
    else if (k < 2 * F) v = x[(size_t)n * F + (k - F)];
    Xcat[idx] = f2bf(v);
}

// --------------------- pack [Wl|Wr] into WMMA B layout ---------------------
// B tile (32x16 bf16, wave32): lane L -> column n = L&15;
//   lanes 0-15 hold K=0..7 & 16..23, lanes 16-31 hold K=8..15 & 24..31 (VGPR j = pair).
// Bpack[((nt*Ksteps + kt)*32 + lane)*8 + vg] = u32 bf16 pair.
__global__ __launch_bounds__(256) void pack_weights_kernel(
    const float* __restrict__ Wl, const float* __restrict__ Wr,
    unsigned* __restrict__ Bpack, int Kin, int Kpad, int Hout, int Ntiles) {
    int Ksteps = Kpad / 32;
    int total = Ntiles * Ksteps * 32 * 8;
    int idx = blockIdx.x * blockDim.x + threadIdx.x;
    if (idx >= total) return;
    int vg   = idx & 7;
    int lane = (idx >> 3) & 31;
    int tk   = idx >> 8;              // nt*Ksteps + kt
    int kt   = tk % Ksteps;
    int nt   = tk / Ksteps;
    int n    = nt * 16 + (lane & 15); // output row of Wcat
    int kloc = ((vg >> 2) ? 16 : 0) + ((lane >> 4) ? 8 : 0) + 2 * (vg & 3);
    unsigned out = 0;
    for (int h = 0; h < 2; ++h) {
        int k = kt * 32 + kloc + h;
        float w = 0.0f;
        if (n < Hout) {
            if (k < Kin)           w = Wl[(size_t)n * Kin + k];
            else if (k < 2 * Kin)  w = Wr[(size_t)n * Kin + (k - Kin)];
        }
        out |= ((unsigned)f2bf(w)) << (16 * h);
    }
    Bpack[idx] = out;
}

// ------------------------------ WMMA GEMM ----------------------------------
// out[M=16 tile][N=16 tile] = Xcat(bf16) @ Wcat^T(bf16) + bias, f32 accum.
// One wave per 16x16 tile, 8 waves per block. EXEC all-1s at every WMMA.
__global__ __launch_bounds__(256) void sage_gemm_kernel(
    const unsigned short* __restrict__ Xcat,   // [N, Kpad] bf16
    const unsigned* __restrict__ Bpack,        // packed B operand
    const float* __restrict__ bias,            // [Hout]
    float* __restrict__ out,                   // [N, Hout]
    int Mtiles, int Kpad, int Hout, int Ntiles) {
    int wave = threadIdx.x >> 5;
    int lane = threadIdx.x & 31;
    int tile = blockIdx.x * 8 + wave;
    if (tile >= Mtiles * Ntiles) return;       // whole-wave uniform exit
    int mt = tile / Ntiles;
    int nt = tile - mt * Ntiles;
    int half = lane >> 4;                      // 0: lanes 0-15, 1: lanes 16-31
    int lid  = lane & 15;
    int Ksteps = Kpad / 32;

    const unsigned short* xr = Xcat + (size_t)(mt * 16 + lid) * Kpad;
    const unsigned* bp = Bpack + ((size_t)(nt * Ksteps) * 32 + lane) * 8;

    v8f c = {};
    for (int kt = 0; kt < Ksteps; ++kt) {
        // A 16x32 bf16: lanes 0-15 -> K 0..7 & 16..23, lanes 16-31 -> K 8..15 & 24..31
        union { unsigned u[8]; v16bf v; } a;
        int kb = kt * 32 + half * 8;
        #pragma unroll
        for (int i = 0; i < 4; ++i) {
            a.u[i]     = *(const unsigned*)(xr + kb + 2 * i);
            a.u[4 + i] = *(const unsigned*)(xr + kb + 16 + 2 * i);
        }
        // B 32x16 bf16: 8 consecutive u32 per lane (prepacked)
        union { unsigned u[8]; v16bf v; } b;
        const unsigned* bk = bp + (size_t)kt * 32 * 8;
        #pragma unroll
        for (int i = 0; i < 8; ++i) b.u[i] = bk[i];

        c = __builtin_amdgcn_wmma_f32_16x16x32_bf16(
                /*neg_a=*/false, a.v, /*neg_b=*/false, b.v,
                /*c_mod=*/(short)0, c, /*reuse_a=*/false, /*reuse_b=*/false);
    }

    // C/D layout: VGPR r -> M = half*8 + r, N = lid
    int n = nt * 16 + lid;
    if (n < Hout) {
        float bv = bias[n];
        int mbase = mt * 16 + half * 8;
        #pragma unroll
        for (int r = 0; r < 8; ++r)
            out[(size_t)(mbase + r) * Hout + n] = c[r] + bv;
    }
}

// --------------------------- BatchNorm reduce ------------------------------
__global__ __launch_bounds__(256) void bn_reduce_kernel(
    const float* __restrict__ h, float* __restrict__ stats, int N, int H) {
    int t = threadIdx.x;
    int rows = 256 / H;                 // H is 32 or 64
    int col = t % H;
    int rlane = t / H;
    float s = 0.0f, s2 = 0.0f;
    for (int r = blockIdx.x * rows + rlane; r < N; r += gridDim.x * rows) {
        float v = h[(size_t)r * H + col];
        s += v; s2 += v * v;
    }
    __shared__ float sm[512];
    sm[t] = s; sm[256 + t] = s2;
    __syncthreads();
    if (rlane == 0) {
        for (int i = 1; i < rows; ++i) {
            s  += sm[i * H + col];
            s2 += sm[256 + i * H + col];
        }
        atomicAdd(&stats[col], s);
        atomicAdd(&stats[H + col], s2);
    }
}

__global__ __launch_bounds__(256) void bn_apply_relu_kernel(
    float* __restrict__ h, const float* __restrict__ stats,
    const float* __restrict__ g, const float* __restrict__ beta,
    int N, int H) {
    int idx = blockIdx.x * blockDim.x + threadIdx.x;
    if (idx >= N * H) return;
    int c = idx % H;
    float invN = 1.0f / (float)N;
    float mu = stats[c] * invN;
    float var = stats[H + c] * invN - mu * mu;
    float v = (h[idx] - mu) * rsqrtf(var + 1e-5f) * g[c] + beta[c];
    h[idx] = v > 0.0f ? v : 0.0f;
}

// ------------------------------ log_softmax --------------------------------
__global__ __launch_bounds__(256) void logsoftmax2_kernel(
    const float* __restrict__ h, float* __restrict__ out, int N) {
    int n = blockIdx.x * blockDim.x + threadIdx.x;
    if (n >= N) return;
    float a = h[n * 2], b = h[n * 2 + 1];
    float m = fmaxf(a, b);
    float lse = m + logf(expf(a - m) + expf(b - m));
    out[n * 2]     = a - lse;
    out[n * 2 + 1] = b - lse;
}

// ---------------------------------------------------------------------------
static inline int cdiv(int a, int b) { return (a + b - 1) / b; }

extern "C" void kernel_launch(void* const* d_in, const int* in_sizes, int n_in,
                              void* d_out, int out_size, void* d_ws, size_t ws_size,
                              hipStream_t stream) {
    const float* x    = (const float*)d_in[0];
    const int*   ei   = (const int*)d_in[1];
    const float* W1l  = (const float*)d_in[2];
    const float* b1l  = (const float*)d_in[3];
    const float* W1r  = (const float*)d_in[4];
    const float* g1   = (const float*)d_in[5];
    const float* be1  = (const float*)d_in[6];
    const float* W2l  = (const float*)d_in[7];
    const float* b2l  = (const float*)d_in[8];
    const float* W2r  = (const float*)d_in[9];
    const float* g2   = (const float*)d_in[10];
    const float* be2  = (const float*)d_in[11];
    const float* W3l  = (const float*)d_in[12];
    const float* b3l  = (const float*)d_in[13];
    const float* W3r  = (const float*)d_in[14];

    const int F_IN = 15, H1 = 64, H2 = 32, OUT = 2;
    const int N = in_sizes[0] / F_IN;
    const int E = in_sizes[1] / 2;
    const int Mtiles = cdiv(N, 16);
    const int* src = ei;
    const int* dst = ei + E;

    // ---- workspace carve-up (all 4-byte aligned) ----
    float* deg            = (float*)d_ws;                       // N
    float* agg            = deg + N;                            // N*64
    unsigned short* Xcat  = (unsigned short*)(agg + (size_t)N * 64);   // N*128 u16
    float* h1             = (float*)(Xcat + (size_t)N * 128);   // N*64
    float* h2             = h1 + (size_t)N * 64;                // N*32
    float* h3             = h2 + (size_t)N * 32;                // N*2
    unsigned* Bpack       = (unsigned*)(h3 + (size_t)N * 2);    // 4096 u32
    float* stats          = (float*)(Bpack + 4096);             // 128

    float* out = (float*)d_out;

    // ---- degrees (shared by all layers) ----
    hipMemsetAsync(deg, 0, (size_t)N * sizeof(float), stream);
    degree_kernel<<<cdiv(E, 256), 256, 0, stream>>>(dst, deg, E);

    // ======================= Layer 1: 15 -> 64 =======================
    {
        const int F = F_IN, Kpad = 32, Hout = H1, Ntiles = cdiv(Hout, 16);
        const int nchunk = cdiv(F, 8);
        hipMemsetAsync(agg, 0, (size_t)N * F * sizeof(float), stream);
        aggregate_kernel<<<cdiv(E * nchunk, 256), 256, 0, stream>>>(
            x, src, dst, agg, E, F, nchunk);
        build_xcat_kernel<<<cdiv(N * Kpad, 256), 256, 0, stream>>>(
            agg, deg, x, Xcat, N, F, Kpad);
        pack_weights_kernel<<<cdiv(Ntiles * (Kpad / 32) * 256, 256), 256, 0, stream>>>(
            W1l, W1r, Bpack, F, Kpad, Hout, Ntiles);
        sage_gemm_kernel<<<cdiv(Mtiles * Ntiles, 8), 256, 0, stream>>>(
            Xcat, Bpack, b1l, h1, Mtiles, Kpad, Hout, Ntiles);
        hipMemsetAsync(stats, 0, 2 * Hout * sizeof(float), stream);
        bn_reduce_kernel<<<256, 256, 0, stream>>>(h1, stats, N, Hout);
        bn_apply_relu_kernel<<<cdiv(N * Hout, 256), 256, 0, stream>>>(
            h1, stats, g1, be1, N, Hout);
    }

    // ======================= Layer 2: 64 -> 32 =======================
    {
        const int F = H1, Kpad = 128, Hout = H2, Ntiles = cdiv(Hout, 16);
        const int nchunk = cdiv(F, 8);
        hipMemsetAsync(agg, 0, (size_t)N * F * sizeof(float), stream);
        aggregate_kernel<<<cdiv(E * nchunk, 256), 256, 0, stream>>>(
            h1, src, dst, agg, E, F, nchunk);
        build_xcat_kernel<<<cdiv(N * Kpad, 256), 256, 0, stream>>>(
            agg, deg, h1, Xcat, N, F, Kpad);
        pack_weights_kernel<<<cdiv(Ntiles * (Kpad / 32) * 256, 256), 256, 0, stream>>>(
            W2l, W2r, Bpack, F, Kpad, Hout, Ntiles);
        sage_gemm_kernel<<<cdiv(Mtiles * Ntiles, 8), 256, 0, stream>>>(
            Xcat, Bpack, b2l, h2, Mtiles, Kpad, Hout, Ntiles);
        hipMemsetAsync(stats, 0, 2 * Hout * sizeof(float), stream);
        bn_reduce_kernel<<<256, 256, 0, stream>>>(h2, stats, N, Hout);
        bn_apply_relu_kernel<<<cdiv(N * Hout, 256), 256, 0, stream>>>(
            h2, stats, g2, be2, N, Hout);
    }

    // ======================= Layer 3: 32 -> 2 ========================
    {
        const int F = H2, Kpad = 64, Hout = OUT, Ntiles = 1;
        const int nchunk = cdiv(F, 8);
        hipMemsetAsync(agg, 0, (size_t)N * F * sizeof(float), stream);
        aggregate_kernel<<<cdiv(E * nchunk, 256), 256, 0, stream>>>(
            h2, src, dst, agg, E, F, nchunk);
        build_xcat_kernel<<<cdiv(N * Kpad, 256), 256, 0, stream>>>(
            agg, deg, h2, Xcat, N, F, Kpad);
        pack_weights_kernel<<<cdiv(Ntiles * (Kpad / 32) * 256, 256), 256, 0, stream>>>(
            W3l, W3r, Bpack, F, Kpad, Hout, Ntiles);
        sage_gemm_kernel<<<cdiv(Mtiles * Ntiles, 8), 256, 0, stream>>>(
            Xcat, Bpack, b3l, h3, Mtiles, Kpad, Hout, Ntiles);
    }

    // ---- final log_softmax over 2 classes ----
    logsoftmax2_kernel<<<cdiv(N, 256), 256, 0, stream>>>(h3, out, N);
}
